// SwinTransformerPhysics_23441931502158
// MI455X (gfx1250) — matrix operand order repaired and checked by reading, not compiled, not run on hardware
//
#include <hip/hip_runtime.h>
#include <hip/hip_bf16.h>
#include <math.h>

typedef __attribute__((ext_vector_type(16))) _Float16 v16h;
typedef __attribute__((ext_vector_type(8)))  _Float16 v8h;
typedef __attribute__((ext_vector_type(8)))  float    v8f;

#define DEV __device__ __forceinline__

// ---------------- CDNA5 async global->LDS copy (ASYNCcnt path) ----------------
DEV void async_ld128(void* lds, const void* gptr) {
  unsigned l = (unsigned)(uintptr_t)lds;               // LDS byte offset (low 32 bits)
  unsigned long long ga = (unsigned long long)(uintptr_t)gptr;
  asm volatile("global_load_async_to_lds_b128 %0, %1, off" :: "v"(l), "v"(ga) : "memory");
}
DEV void wait_async0() { asm volatile("s_wait_asynccnt 0x0" ::: "memory"); }

// ---------------- WMMA fragment helpers (CDNA5 wave32 layouts) ----------------
// A (16x32 f16): lane = row M (lane%16); element e<8  -> K = k0 + 8*(lane/16) + e
//                                        element e>=8 -> K = k0 + 16 + 8*(lane/16) + (e-8)
DEV v16h frag_a(const _Float16* __restrict__ row, int k0, int hi8) {
  v8h lo = *(const v8h*)(row + k0 + hi8);
  v8h hi = *(const v8h*)(row + k0 + 16 + hi8);
  v16h a;
#pragma unroll
  for (int e = 0; e < 8; ++e) { a[e] = lo[e]; a[e + 8] = hi[e]; }
  return a;
}
// B (32x16 f16): lane = col N (lane%16); element e<8 -> K = k0 + 8*(lane/16) + e
//                                        element e>=8 -> K = k0 + 16 + 8*(lane/16) + (e-8)
DEV v16h frag_b(const _Float16* __restrict__ rowN, int k0, int hi16) {
  v8h lo = *(const v8h*)(rowN + k0 + hi16 / 2);
  v8h hi = *(const v8h*)(rowN + k0 + 16 + hi16 / 2);
  v16h b;
#pragma unroll
  for (int e = 0; e < 8; ++e) { b[e] = lo[e]; b[e + 8] = hi[e]; }
  return b;
}
DEV v8f wmma_f16(v16h a, v16h b, v8f c) {
  return __builtin_amdgcn_wmma_f32_16x16x32_f16(false, a, false, b, (short)0, c, false, false);
}

// ---------------- problem constants ----------------
#define Bn    4
#define Cn    4
#define Dn    96
#define NSn   12
#define HINn  128
#define TZn   64
#define PZn   32
#define NVOX  (PZn*PZn*PZn)            // 32768
#define NTOKA (Bn*NVOX)                // 131072

// f16 weight segment offsets (halfs)
#define O_QKV 0
#define O_ATT 27648
#define O_ENC 36864
#define O_PE  46080
#define O_W1  52224
#define O_W2  301056
#define O_REC 549888
#define N_W16 562176

// ---------------- K0: weights -> f16, reordered ----------------
__global__ void k_prep(const float* __restrict__ qkv_w, const float* __restrict__ attn_w,
                       const float* __restrict__ enc_w, const float* __restrict__ pe_w,
                       const float* __restrict__ w1, const float* __restrict__ w2,
                       const float* __restrict__ rec_w, _Float16* __restrict__ W) {
  int j = blockIdx.x * 256 + threadIdx.x;
  if (j < 27648) { W[O_QKV + j] = (_Float16)qkv_w[j]; return; } j -= 27648;
  if (j < 9216)  { W[O_ATT + j] = (_Float16)attn_w[j]; return; } j -= 9216;
  if (j < 9216)  { W[O_ENC + j] = (_Float16)enc_w[j]; return; }  j -= 9216;
  if (j < 6144)  { W[O_PE  + j] = (_Float16)pe_w[j]; return; }   j -= 6144;
  if (j < 248832) { int tap = j / 9216, oi = j % 9216;
                    W[O_W1 + j] = (_Float16)w1[(size_t)oi * 27 + tap]; return; } j -= 248832;
  if (j < 248832) { int tap = j / 9216, oi = j % 9216;
                    W[O_W2 + j] = (_Float16)w2[(size_t)oi * 27 + tap]; return; } j -= 248832;
  if (j < 12288) { // WREC[n=(c,d,e,f)][k=i] = rec_w[i][c][d][e][f]
    int n = j / 96, ii = j % 96;
    int cc = n >> 5, dd = (n >> 4) & 1, ee = (n >> 2) & 3, ff = n & 3;
    W[O_REC + j] = (_Float16)rec_w[((((size_t)ii * 4 + cc) * 2 + dd) * 4 + ee) * 4 + ff];
  }
}

// ---------------- K1: baseline = z_interp(sparse_obs, pos, 64) -> d_out ----------------
__global__ void k_baseline(const float* __restrict__ x, const long long* __restrict__ sidx,
                           float* __restrict__ out) {
  size_t i = (size_t)blockIdx.x * 256 + threadIdx.x;
  const size_t total = (size_t)Bn * Cn * TZn * HINn * HINn;
  if (i >= total) return;
  int w = i % HINn; size_t r = i / HINn;
  int h = r % HINn; r /= HINn;
  int z = r % TZn;  r /= TZn;
  int c = r % Cn;   int b = (int)(r / Cn);
  float pos[NSn];
#pragma unroll
  for (int s = 0; s < NSn; ++s) pos[s] = (float)sidx[s];
  float zq = (float)z;
  int right = NSn;
#pragma unroll
  for (int s = 0; s < NSn; ++s) { if (right == NSn && pos[s] >= zq) right = s; }
  int lf = min(max(right - 1, 0), NSn - 1);
  int rc = min(max(right, 0), NSn - 1);
  float lp = pos[lf], rp = pos[rc], dd = rp - lp;
  float wg = dd > 0.f ? (zq - lp) / dd : 0.5f;
  size_t bL = ((((size_t)b * Cn + c) * NSn + lf) * HINn + h) * HINn + w;
  size_t bR = ((((size_t)b * Cn + c) * NSn + rc) * HINn + h) * HINn + w;
  out[i] = (1.f - wg) * x[bL] + wg * x[bR];
}

// ---------------- K2: patch embed (WMMA, K=64) + LayerNorm -> f0 token-major ----------------
__global__ void __launch_bounds__(128)
k_patch_ln(const float* __restrict__ xin, const _Float16* __restrict__ Wpe,
           const float* __restrict__ pe_b, const float* __restrict__ pe_g,
           const float* __restrict__ pe_beta, float* __restrict__ f0) {
  __shared__ __align__(16) _Float16 A[64][64];
  __shared__ float O[64][96];
  const int tid = threadIdx.x, wave = tid >> 5, lane = tid & 31;
  const int hi8 = (lane >> 4) * 8, hi16 = (lane >> 4) * 16, nloc = lane & 15, mhi = (lane >> 4) << 3;
  const int t0 = blockIdx.x * 64;
  const int b = t0 / (NSn * 1024);
  const int s = (t0 / 1024) % NSn;
  for (int e = tid; e < 64 * 64; e += 128) {
    int tt = e >> 6, k = e & 63;
    int t = t0 + tt;
    int hh = (t >> 5) & 31, ww = t & 31;
    int cch = k >> 4, pp = (k >> 2) & 3, qq = k & 3;
    size_t gi = ((((size_t)b * Cn + cch) * NSn + s) * HINn + (hh * 4 + pp)) * HINn + (ww * 4 + qq);
    A[tt][k] = (_Float16)xin[gi];
  }
  __syncthreads();
  for (int t = wave; t < 24; t += 4) {
    int mi = t / 6, ni = t % 6;
    v8f acc = {};
#pragma unroll
    for (int k0 = 0; k0 < 64; k0 += 32)
      acc = wmma_f16(frag_a(&A[mi * 16 + (lane & 15)][0], k0, hi8),
                     frag_b(Wpe + (size_t)(ni * 16 + nloc) * 64, k0, hi16), acc);
    int n = ni * 16 + nloc;
#pragma unroll
    for (int r = 0; r < 8; ++r) O[mi * 16 + r + mhi][n] = acc[r];
  }
  __syncthreads();
  if (tid < 64) {
    float mv = 0.f;
    for (int n = 0; n < 96; ++n) mv += O[tid][n] + pe_b[n];
    mv *= (1.f / 96.f);
    float vv = 0.f;
    for (int n = 0; n < 96; ++n) { float d = O[tid][n] + pe_b[n] - mv; vv += d * d; }
    vv *= (1.f / 96.f);
    float rs = rsqrtf(vv + 1e-5f);
    for (int n = 0; n < 96; ++n)
      f0[(size_t)(t0 + tid) * 96 + n] = (O[tid][n] + pe_b[n] - mv) * rs * pe_g[n] + pe_beta[n];
  }
}

// ---------------- K3: f = z_interp(f0, pos/2, 32) + enc @ enc_w^T (WMMA) ----------------
__global__ void __launch_bounds__(128)
k_zinterp_enc(const float* __restrict__ f0, const float* __restrict__ enc,
              const _Float16* __restrict__ Wenc, const float* __restrict__ enc_b,
              const long long* __restrict__ sidx, float* __restrict__ f) {
  __shared__ __align__(16) _Float16 A[64][96];
  __shared__ float O[64][96];
  const int tid = threadIdx.x, wave = tid >> 5, lane = tid & 31;
  const int hi8 = (lane >> 4) * 8, hi16 = (lane >> 4) * 16, nloc = lane & 15, mhi = (lane >> 4) << 3;
  const int t0 = blockIdx.x * 64;
  const int b = t0 >> 15;
  const int z = (t0 >> 10) & 31;
  const int hw0 = t0 & 1023;
  for (int e = tid; e < 64 * 96; e += 128) {          // coalesced over tokens per channel
    int ch = e >> 6, i = e & 63;
    size_t gi = ((size_t)b * 96 + ch) * NVOX + ((size_t)z * 1024 + hw0 + i);
    A[i][ch] = (_Float16)enc[gi];
  }
  __syncthreads();
  for (int t = wave; t < 24; t += 4) {
    int mi = t / 6, ni = t % 6;
    v8f acc = {};
#pragma unroll
    for (int k0 = 0; k0 < 96; k0 += 32)
      acc = wmma_f16(frag_a(&A[mi * 16 + (lane & 15)][0], k0, hi8),
                     frag_b(Wenc + (size_t)(ni * 16 + nloc) * 96, k0, hi16), acc);
    int n = ni * 16 + nloc;
#pragma unroll
    for (int r = 0; r < 8; ++r) O[mi * 16 + r + mhi][n] = acc[r] + enc_b[n];
  }
  __syncthreads();
  float pos[NSn];
#pragma unroll
  for (int s = 0; s < NSn; ++s) pos[s] = 0.5f * (float)sidx[s];
  float zq = (float)z;
  int right = NSn;
#pragma unroll
  for (int s = 0; s < NSn; ++s) { if (right == NSn && pos[s] >= zq) right = s; }
  int lf = min(max(right - 1, 0), NSn - 1);
  int rc = min(max(right, 0), NSn - 1);
  float lp = pos[lf], rp = pos[rc], dd = rp - lp;
  float wg = dd > 0.f ? (zq - lp) / dd : 0.5f;
  for (int e = tid; e < 64 * 96; e += 128) {
    int i = e / 96, n = e % 96;
    int hw = hw0 + i;
    size_t iL = (((size_t)b * NSn + lf) * 1024 + hw) * 96 + n;
    size_t iR = (((size_t)b * NSn + rc) * 1024 + hw) * 96 + n;
    f[((size_t)t0 + i) * 96 + n] = (1.f - wg) * f0[iL] + wg * f0[iR] + O[i][n];
  }
}

// ---------------- K4: fused window attention (one 4x4x4 window per block) ----------------
// Output g is f16 token-major (consumed only as WMMA A operand downstream).
__global__ void __launch_bounds__(128)
k_attn(const float* __restrict__ f, const _Float16* __restrict__ Wqkv,
       const _Float16* __restrict__ Watt,
       const float* __restrict__ ln_g, const float* __restrict__ ln_b,
       const float* __restrict__ qkv_b, const float* __restrict__ attn_b,
       const float* __restrict__ rpb, _Float16* __restrict__ g) {
  __shared__ __align__(16) _Float16 SC[64][96];       // residual input (f16)
  __shared__ __align__(16) _Float16 XN[64][96];       // LN'd tokens; reused as scores/P (64x64)
  __shared__ __align__(16) _Float16 Qs[3][64][32];    // scaled Q; reused as O per head
  __shared__ __align__(16) _Float16 Ks[3][64][32];
  __shared__ __align__(16) _Float16 Vt[3][32][64];    // V transposed (d-major)
  const int tid = threadIdx.x, wave = tid >> 5, lane = tid & 31;
  const int hi8 = (lane >> 4) * 8, hi16 = (lane >> 4) * 16, nloc = lane & 15, mhi = (lane >> 4) << 3;
  const int wid = blockIdx.x;
  const int b = wid >> 9;
  const int iz = (wid >> 6) & 7, ih = (wid >> 3) & 7, iw = wid & 7;
  auto voxIdx = [&](int tt) -> size_t {
    int lz = tt >> 4, lh = (tt >> 2) & 3, lw = tt & 3;
    return (((size_t)b * 32 + (iz * 4 + lz)) * 32 + (ih * 4 + lh)) * 32 + (iw * 4 + lw);
  };
  for (int e = tid; e < 64 * 96; e += 128) {
    int tt = e / 96, ch = e % 96;
    SC[tt][ch] = (_Float16)f[voxIdx(tt) * 96 + ch];
  }
  __syncthreads();
  if (tid < 64) {                               // LayerNorm
    float mv = 0.f;
    for (int ch = 0; ch < 96; ++ch) mv += (float)SC[tid][ch];
    mv *= (1.f / 96.f);
    float vv = 0.f;
    for (int ch = 0; ch < 96; ++ch) { float d = (float)SC[tid][ch] - mv; vv += d * d; }
    vv *= (1.f / 96.f);
    float rs = rsqrtf(vv + 1e-5f);
    for (int ch = 0; ch < 96; ++ch)
      XN[tid][ch] = (_Float16)(((float)SC[tid][ch] - mv) * rs * ln_g[ch] + ln_b[ch]);
  }
  __syncthreads();
  // QKV GEMM: 64x288, K=96 -> 72 tiles
  for (int t = wave; t < 72; t += 4) {
    int mi = t / 18, ni = t % 18;
    v8f acc = {};
#pragma unroll
    for (int k0 = 0; k0 < 96; k0 += 32)
      acc = wmma_f16(frag_a(&XN[mi * 16 + (lane & 15)][0], k0, hi8),
                     frag_b(Wqkv + (size_t)(ni * 16 + nloc) * 96, k0, hi16), acc);
    int n = ni * 16 + nloc;
    int which = n / 96, hd = (n % 96) / 32, dl = n % 32;
#pragma unroll
    for (int r = 0; r < 8; ++r) {
      int m = mi * 16 + r + mhi;
      float v = acc[r] + qkv_b[n];
      if (which == 0)      Qs[hd][m][dl] = (_Float16)(v * 0.17677669529663687f); // *dh^-0.5
      else if (which == 1) Ks[hd][m][dl] = (_Float16)v;
      else                 Vt[hd][dl][m] = (_Float16)v;
    }
  }
  __syncthreads();
  _Float16* ATT = &XN[0][0];                    // reuse as 64x64 scores / P
  for (int hd = 0; hd < 3; ++hd) {
    for (int t = wave; t < 16; t += 4) {        // S = Q K^T (K=32)
      int mi = t / 4, ni = t % 4;
      v8f acc = {};
      acc = wmma_f16(frag_a(&Qs[hd][mi * 16 + (lane & 15)][0], 0, hi8),
                     frag_b(&Ks[hd][ni * 16 + nloc][0], 0, hi16), acc);
      int n = ni * 16 + nloc;
#pragma unroll
      for (int r = 0; r < 8; ++r) ATT[(mi * 16 + r + mhi) * 64 + n] = (_Float16)acc[r];
    }
    __syncthreads();
    if (tid < 64) {                             // softmax with relative position bias
      int qz = tid >> 4, qh = (tid >> 2) & 3, qw = tid & 3;
      float mx = -1e30f;
      for (int m = 0; m < 64; ++m) {
        int kz = m >> 4, kh = (m >> 2) & 3, kw = m & 3;
        int ridx = (qz - kz + 3) * 49 + (qh - kh + 3) * 7 + (qw - kw + 3);
        float v = (float)ATT[tid * 64 + m] + rpb[ridx * 3 + hd];
        mx = fmaxf(mx, v);
      }
      float sum = 0.f;
      for (int m = 0; m < 64; ++m) {
        int kz = m >> 4, kh = (m >> 2) & 3, kw = m & 3;
        int ridx = (qz - kz + 3) * 49 + (qh - kh + 3) * 7 + (qw - kw + 3);
        float v = __expf((float)ATT[tid * 64 + m] + rpb[ridx * 3 + hd] - mx);
        sum += v;
        ATT[tid * 64 + m] = (_Float16)v;
      }
      float inv = 1.f / sum;
      for (int m = 0; m < 64; ++m)
        ATT[tid * 64 + m] = (_Float16)((float)ATT[tid * 64 + m] * inv);
    }
    __syncthreads();
    for (int t = wave; t < 8; t += 4) {         // O = P V (K=64) -> store into Qs[hd]
      int mi = t / 2, ni = t % 2;
      v8f acc = {};
#pragma unroll
      for (int k0 = 0; k0 < 64; k0 += 32)
        acc = wmma_f16(frag_a(ATT + (size_t)(mi * 16 + (lane & 15)) * 64, k0, hi8),
                       frag_b(&Vt[hd][ni * 16 + nloc][0], k0, hi16), acc);
      int n = ni * 16 + nloc;
#pragma unroll
      for (int r = 0; r < 8; ++r) Qs[hd][mi * 16 + r + mhi][n] = (_Float16)acc[r];
    }
    __syncthreads();
  }
  // output projection + residual: xout = sc + O @ attn_w^T + b
  const _Float16* Ob = &Qs[0][0][0];            // [head][64][32]; k = head*32+d
  for (int t = wave; t < 24; t += 4) {
    int mi = t / 6, ni = t % 6;
    int tokm = mi * 16 + (lane & 15);
    v8f acc = {};
#pragma unroll
    for (int k0 = 0; k0 < 96; k0 += 32) {
      int k1 = k0 + hi8, k2 = k0 + 16 + hi8;    // each 8-chunk stays within one head block
      v8h lo = *(const v8h*)(Ob + ((size_t)(k1 >> 5) * 64 + tokm) * 32 + (k1 & 31));
      v8h hi = *(const v8h*)(Ob + ((size_t)(k2 >> 5) * 64 + tokm) * 32 + (k2 & 31));
      v16h a;
#pragma unroll
      for (int e = 0; e < 8; ++e) { a[e] = lo[e]; a[e + 8] = hi[e]; }
      acc = wmma_f16(a, frag_b(Watt + (size_t)(ni * 16 + nloc) * 96, k0, hi16), acc);
    }
    int n = ni * 16 + nloc;
#pragma unroll
    for (int r = 0; r < 8; ++r) {
      int m = mi * 16 + r + mhi;
      g[voxIdx(m) * 96 + n] = (_Float16)((float)SC[m][n] + acc[r] + attn_b[n]);
    }
  }
}

// ---------------- K5: implicit-GEMM 3x3x3 conv, 96->96 (one (b,z,h) row per block) -----
// f16 input staged into LDS via CDNA5 async global->LDS copies (ASYNCcnt).
__global__ void __launch_bounds__(128)
k_conv3(const _Float16* __restrict__ xin, const _Float16* __restrict__ Wt,
        const float* __restrict__ bias, float* __restrict__ yout) {
  __shared__ __align__(16) _Float16 X[9][34][96];   // 9 (dz,dh) rows, w in [-1,32]
  const int tid = threadIdx.x, wave = tid >> 5, lane = tid & 31;
  const int hi8 = (lane >> 4) * 8, hi16 = (lane >> 4) * 16, nloc = lane & 15;
  const int mhi = (lane >> 4) << 3, mrow = lane & 15;
  const int blk = blockIdx.x;
  const int b = blk >> 10, z = (blk >> 5) & 31, h = blk & 31;
  // 1) zero-fill (SAME padding borders); 29376 halfs = 3672 x 16B
  {
    v8h zz = {};
    for (int e = tid * 8; e < 9 * 34 * 96; e += 128 * 8)
      *(v8h*)(&X[0][0][0] + e) = zz;
  }
  __syncthreads();
  // 2) async-copy interior: per (dz,dh) row, 32 voxels x 96 ch = 384 x 16B chunks
  for (int c = tid; c < 9 * 384; c += 128) {
    int r = c / 384, q = c % 384;
    int wv = q / 12;                 // voxel w (0..31)
    int ch0 = (q % 12) * 8;          // half offset within channel dim
    int zs = z + r / 3 - 1, hs = h + r % 3 - 1;
    if (zs >= 0 && zs < 32 && hs >= 0 && hs < 32) {
      const _Float16* gp = xin + ((((size_t)b * 32 + zs) * 32 + hs) * 32 + wv) * 96 + ch0;
      async_ld128(&X[r][wv + 1][ch0], gp);
    }
  }
  wait_async0();
  __syncthreads();
  v8f acc0 = {}, acc1 = {}, acc2 = {};
  for (int tap = 0; tap < 27; ++tap) {
    const _Float16* Wtap = Wt + (size_t)tap * 9216;
    __builtin_prefetch(Wt + (size_t)((tap + 1) % 27) * 9216, 0, 1); // global_prefetch_b8
    int dw = tap % 3;
    int r9 = tap / 3;                            // dz*3+dh
#pragma unroll
    for (int ti = 0; ti < 3; ++ti) {
      int t = wave + ti * 4;                     // 12 tiles over 4 waves
      int mi = t / 6, ni = t % 6;
      const _Float16* arow = &X[r9][mi * 16 + mrow + dw][0];
      const _Float16* brow = Wtap + (size_t)(ni * 16 + nloc) * 96;
      v8f a = (ti == 0) ? acc0 : (ti == 1) ? acc1 : acc2;
#pragma unroll
      for (int k0 = 0; k0 < 96; k0 += 32)
        a = wmma_f16(frag_a(arow, k0, hi8), frag_b(brow, k0, hi16), a);
      if (ti == 0) acc0 = a; else if (ti == 1) acc1 = a; else acc2 = a;
    }
  }
#pragma unroll
  for (int ti = 0; ti < 3; ++ti) {
    int t = wave + ti * 4;
    int mi = t / 6, ni = t % 6;
    int n = ni * 16 + nloc;
    v8f a = (ti == 0) ? acc0 : (ti == 1) ? acc1 : acc2;
#pragma unroll
    for (int r = 0; r < 8; ++r) {
      int m = mi * 16 + r + mhi;
      yout[((((size_t)b * 32 + z) * 32 + h) * 32 + m) * 96 + n] = a[r] + bias[n];
    }
  }
}

// ---------------- GroupNorm (8 groups) two-stage stats + apply+GELU -> f16 ----------------
__global__ void k_gnstat1(const float* __restrict__ x, float* __restrict__ part) {
  __shared__ float ssum[256], ssq[256];
  int bid = blockIdx.x;
  int blk = bid & 15, bg = bid >> 4;             // bg = b*8+g in [0,32)
  int b = bg >> 3, gg = bg & 7;
  float s = 0.f, q = 0.f;
  for (int e = threadIdx.x; e < 2048 * 12; e += 256) {
    int vox = blk * 2048 + e / 12;
    int ch = gg * 12 + e % 12;
    float v = x[((size_t)b * NVOX + vox) * 96 + ch];
    s += v; q += v * v;
  }
  ssum[threadIdx.x] = s; ssq[threadIdx.x] = q;
  __syncthreads();
  for (int st = 128; st > 0; st >>= 1) {
    if (threadIdx.x < st) { ssum[threadIdx.x] += ssum[threadIdx.x + st];
                            ssq[threadIdx.x]  += ssq[threadIdx.x + st]; }
    __syncthreads();
  }
  if (threadIdx.x == 0) { part[bid * 2] = ssum[0]; part[bid * 2 + 1] = ssq[0]; }
}
__global__ void k_gnstat2(const float* __restrict__ part, float* __restrict__ stats) {
  int bg = threadIdx.x;
  if (bg >= 32) return;
  float s = 0.f, q = 0.f;
  for (int i = 0; i < 16; ++i) { s += part[(bg * 16 + i) * 2]; q += part[(bg * 16 + i) * 2 + 1]; }
  const float N = 12.f * 32768.f;
  float m = s / N, v = q / N - m * m;
  stats[bg * 2] = m; stats[bg * 2 + 1] = rsqrtf(v + 1e-5f);
}
__global__ void k_gnapply(const float* __restrict__ x, const float* __restrict__ stats,
                          const float* __restrict__ gg, const float* __restrict__ bb,
                          _Float16* __restrict__ y16) {
  size_t i = (size_t)blockIdx.x * 256 + threadIdx.x;
  if (i >= (size_t)NTOKA * 96) return;
  int ch = i % 96;
  int b = (int)((i / 96) >> 15);
  int grp = ch / 12;
  float m = stats[(b * 8 + grp) * 2], rs = stats[(b * 8 + grp) * 2 + 1];
  float y = (x[i] - m) * rs * gg[ch] + bb[ch];
  y16[i] = (_Float16)(y * 0.5f * (1.f + erff(y * 0.70710678118654752f)));  // exact GELU
}

// ---------------- K6: recon GEMM (K=96, N=128) scatter-add onto baseline ----------------
__global__ void __launch_bounds__(128)
k_recon(const _Float16* __restrict__ fin, const _Float16* __restrict__ Wrec,
        const float* __restrict__ rec_b, float* __restrict__ out) {
  __shared__ __align__(16) _Float16 A[64][96];
  const int tid = threadIdx.x, wave = tid >> 5, lane = tid & 31;
  const int hi8 = (lane >> 4) * 8, hi16 = (lane >> 4) * 16, nloc = lane & 15, mhi = (lane >> 4) << 3;
  const int t0 = blockIdx.x * 64;
  // async-copy the 64x96 f16 A tile (768 x 16B chunks)
  for (int c = tid; c < (64 * 96) / 8; c += 128)
    async_ld128(&A[0][0] + c * 8, fin + (size_t)t0 * 96 + c * 8);
  wait_async0();
  __syncthreads();
  for (int t = wave; t < 32; t += 4) {
    int mi = t / 8, ni = t % 8;
    v8f acc = {};
#pragma unroll
    for (int k0 = 0; k0 < 96; k0 += 32)
      acc = wmma_f16(frag_a(&A[mi * 16 + (lane & 15)][0], k0, hi8),
                     frag_b(Wrec + (size_t)(ni * 16 + nloc) * 96, k0, hi16), acc);
    int n = ni * 16 + nloc;
    int cc = n >> 5, dd = (n >> 4) & 1, ee = (n >> 2) & 3, ff = n & 3;
    float bias = rec_b[cc];
#pragma unroll
    for (int r = 0; r < 8; ++r) {
      int tok = t0 + mi * 16 + r + mhi;
      int b = tok >> 15, z = (tok >> 10) & 31, hh = (tok >> 5) & 31, ww = tok & 31;
      size_t oi = (((size_t)(b * 4 + cc) * 64 + (2 * z + dd)) * 128 + (4 * hh + ee)) * 128
                  + (4 * ww + ff);
      out[oi] += acc[r] + bias;
    }
  }
}

// ---------------- host-side orchestration ----------------
extern "C" void kernel_launch(void* const* d_in, const int* in_sizes, int n_in,
                              void* d_out, int out_size, void* d_ws, size_t ws_size,
                              hipStream_t stream) {
  (void)in_sizes; (void)n_in; (void)out_size; (void)ws_size;
  const float* sparse_obs   = (const float*)d_in[0];
  const float* encoder_feat = (const float*)d_in[1];
  const float* pe_w    = (const float*)d_in[2];
  const float* pe_b    = (const float*)d_in[3];
  const float* pe_g    = (const float*)d_in[4];
  const float* pe_beta = (const float*)d_in[5];
  const float* enc_w   = (const float*)d_in[6];
  const float* enc_b   = (const float*)d_in[7];
  const float* ln_g    = (const float*)d_in[8];
  const float* ln_b    = (const float*)d_in[9];
  const float* qkv_w   = (const float*)d_in[10];
  const float* qkv_b   = (const float*)d_in[11];
  const float* attn_w  = (const float*)d_in[12];
  const float* attn_b  = (const float*)d_in[13];
  const float* rpb     = (const float*)d_in[14];
  const float* ref_w1  = (const float*)d_in[15];
  const float* ref_b1  = (const float*)d_in[16];
  const float* gn1_g   = (const float*)d_in[17];
  const float* gn1_b   = (const float*)d_in[18];
  const float* ref_w2  = (const float*)d_in[19];
  const float* ref_b2  = (const float*)d_in[20];
  const float* gn2_g   = (const float*)d_in[21];
  const float* gn2_b   = (const float*)d_in[22];
  const float* rec_w   = (const float*)d_in[23];
  const float* rec_b   = (const float*)d_in[24];
  const long long* sidx = (const long long*)d_in[25];

  char* ws = (char*)d_ws;
  float*     f0  = (float*)(ws);                         // 4,718,592 f32  (18,874,368 B)
  float*     f   = (float*)(ws + 18874368ull);           // 12,582,912 f32 (50,331,648 B)
  float*     t1  = (float*)(ws + 69206016ull);           // 12,582,912 f32
  _Float16*  g16 = (_Float16*)(ws + 119537664ull);       // 12,582,912 f16 (25,165,824 B)
  _Float16*  h16 = (_Float16*)(ws + 144703488ull);       // 12,582,912 f16
  _Float16*  W16 = (_Float16*)(ws + 169869312ull);       // 562,176 f16
  float* part  = (float*)((char*)(W16 + N_W16));         // 1024 f32
  float* stats = part + 1024;                            // 64 f32
  float* outp = (float*)d_out;

  k_prep<<<(N_W16 + 255) / 256, 256, 0, stream>>>(qkv_w, attn_w, enc_w, pe_w,
                                                  ref_w1, ref_w2, rec_w, W16);
  k_baseline<<<65536, 256, 0, stream>>>(sparse_obs, sidx, outp);
  k_patch_ln<<<768, 128, 0, stream>>>(sparse_obs, W16 + O_PE, pe_b, pe_g, pe_beta, f0);
  k_zinterp_enc<<<2048, 128, 0, stream>>>(f0, encoder_feat, W16 + O_ENC, enc_b, sidx, f);
  k_attn<<<2048, 128, 0, stream>>>(f, W16 + O_QKV, W16 + O_ATT,
                                   ln_g, ln_b, qkv_b, attn_b, rpb, g16);
  k_conv3<<<4096, 128, 0, stream>>>(g16, W16 + O_W1, ref_b1, t1);
  k_gnstat1<<<512, 256, 0, stream>>>(t1, part);
  k_gnstat2<<<1, 32, 0, stream>>>(part, stats);
  k_gnapply<<<(12582912 + 255) / 256, 256, 0, stream>>>(t1, stats, gn1_g, gn1_b, h16);
  k_conv3<<<4096, 128, 0, stream>>>(h16, W16 + O_W2, ref_b2, f);
  k_gnstat1<<<512, 256, 0, stream>>>(f, part);
  k_gnstat2<<<1, 32, 0, stream>>>(part, stats);
  k_gnapply<<<(12582912 + 255) / 256, 256, 0, stream>>>(f, stats, gn2_g, gn2_b, h16);
  k_recon<<<2048, 128, 0, stream>>>(h16, W16 + O_REC, rec_b, outp);
}
